// SparseMoELayer_40742059770284
// MI455X (gfx1250) — matrix operand order repaired and checked by reading, not compiled, not run on hardware
//
#include <hip/hip_runtime.h>
#include <hip/hip_bf16.h>

#define D_MODEL   1024
#define D_FF      2816
#define N_TOK     4096
#define N_EXP     8
#define FCHUNK    128
#define TILE_M    16

typedef __attribute__((ext_vector_type(16))) __bf16 v16bf;
typedef __attribute__((ext_vector_type(8)))  float  v8f;

// ---------------------------------------------------------------------------
// Fragment loaders (wave32 WMMA, 16-bit operands; A/B use this code's contiguous k order)
// A 16x32 (MxK): lanes 0-15 -> M=lane, K = k0+0..15;        lanes 16-31 -> M=lane-16, K = k0+16..31
// B 32x16 (KxN): lanes 0-15 -> N=lane, K = k0+0..15;        lanes 16-31 -> N=lane-16, K = k0+16..31
// C/D 16x16:     lanes 0-15 -> N=lane, VGPR r -> M=r;       lanes 16-31 -> N=lane-16, VGPR r -> M=r+8
// ---------------------------------------------------------------------------

__device__ __forceinline__ v16bf lds_a_frag(const __bf16* base, int m, int kb,
                                            int stride, int hsel) {
  // 16 contiguous bf16 (two 16B loads)
  const __bf16* p0 = base + m * stride + kb + hsel * 16;
  const __bf16* p1 = p0 + 8;
  union { uint4 q[2]; v16bf v; } u;
  u.q[0] = *(const uint4*)p0;
  u.q[1] = *(const uint4*)p1;
  return u.v;
}

__device__ __forceinline__ v16bf glb_b_frag_f32(const float* p) {
  // 16 contiguous fp32 -> bf16 fragment
  const float4* q = (const float4*)p;
  float4 a = q[0], b = q[1], c = q[2], d = q[3];
  v16bf v;
  v[0]  = (__bf16)a.x; v[1]  = (__bf16)a.y; v[2]  = (__bf16)a.z; v[3]  = (__bf16)a.w;
  v[4]  = (__bf16)b.x; v[5]  = (__bf16)b.y; v[6]  = (__bf16)b.z; v[7]  = (__bf16)b.w;
  v[8]  = (__bf16)c.x; v[9]  = (__bf16)c.y; v[10] = (__bf16)c.z; v[11] = (__bf16)c.w;
  v[12] = (__bf16)d.x; v[13] = (__bf16)d.y; v[14] = (__bf16)d.z; v[15] = (__bf16)d.w;
  return v;
}

// ---------------------------------------------------------------------------
// Kernel 0: zero output + counters
// ---------------------------------------------------------------------------
__global__ void moe_init(float* out, int out_n, int* cnt, float* rsum) {
  int i = blockIdx.x * blockDim.x + threadIdx.x;
  for (; i < out_n; i += gridDim.x * blockDim.x) out[i] = 0.0f;
  if (blockIdx.x == 0 && threadIdx.x < N_EXP) {
    cnt[threadIdx.x] = 0;
    rsum[threadIdx.x] = 0.0f;
  }
}

// ---------------------------------------------------------------------------
// Kernel 1: gating (softmax + top-2 + dispatch lists + routing mean sums)
// ---------------------------------------------------------------------------
__global__ __launch_bounds__(256)
void moe_gate(const float* __restrict__ x, const float* __restrict__ gate_w,
              int* __restrict__ cnt, int* __restrict__ tok_list,
              float* __restrict__ gate_list, float* __restrict__ rsum) {
  __shared__ float gw[N_EXP][D_MODEL];   // 32 KB
  int tid = threadIdx.x;
  for (int i = tid; i < N_EXP * D_MODEL; i += 256)
    ((float*)gw)[i] = gate_w[i];
  __syncthreads();

  int t = blockIdx.x * 256 + tid;
  if (t >= N_TOK) return;

  float acc[N_EXP];
#pragma unroll
  for (int e = 0; e < N_EXP; ++e) acc[e] = 0.0f;

  const float* xr = x + (size_t)t * D_MODEL;
  for (int c = 0; c < D_MODEL; c += 4) {
    float4 xv = *(const float4*)(xr + c);
#pragma unroll
    for (int e = 0; e < N_EXP; ++e) {
      acc[e] += xv.x * gw[e][c] + xv.y * gw[e][c + 1] +
                xv.z * gw[e][c + 2] + xv.w * gw[e][c + 3];
    }
  }
  // softmax
  float mx = acc[0];
#pragma unroll
  for (int e = 1; e < N_EXP; ++e) mx = fmaxf(mx, acc[e]);
  float s = 0.0f;
#pragma unroll
  for (int e = 0; e < N_EXP; ++e) { acc[e] = __expf(acc[e] - mx); s += acc[e]; }
  float inv = __builtin_amdgcn_rcpf(s);
#pragma unroll
  for (int e = 0; e < N_EXP; ++e) acc[e] *= inv;

  // top-2
  int e1 = 0;
#pragma unroll
  for (int e = 1; e < N_EXP; ++e) if (acc[e] > acc[e1]) e1 = e;
  int e2 = (e1 == 0) ? 1 : 0;
#pragma unroll
  for (int e = 0; e < N_EXP; ++e) if (e != e1 && acc[e] > acc[e2]) e2 = e;

  float norm = __builtin_amdgcn_rcpf(acc[e1] + acc[e2]);
  float g1 = acc[e1] * norm, g2 = acc[e2] * norm;

  int p1 = atomicAdd(&cnt[e1], 1);
  tok_list[e1 * N_TOK + p1] = t;
  gate_list[e1 * N_TOK + p1] = g1;
  int p2 = atomicAdd(&cnt[e2], 1);
  tok_list[e2 * N_TOK + p2] = t;
  gate_list[e2 * N_TOK + p2] = g2;

#pragma unroll
  for (int e = 0; e < N_EXP; ++e) unsafeAtomicAdd(&rsum[e], acc[e]);
}

// ---------------------------------------------------------------------------
// Kernel 2: load-balance loss
// ---------------------------------------------------------------------------
__global__ void moe_loss(const int* cnt, const float* rsum, float* out_loss) {
  float loss = 0.0f;
  for (int e = 0; e < N_EXP; ++e) {
    float f_i = (float)cnt[e] / ((float)(N_TOK * 2) + 1e-6f);
    float I_i = rsum[e] / (float)N_TOK;
    loss += f_i * I_i;
  }
  *out_loss = 0.01f * (float)N_EXP * loss;
}

// ---------------------------------------------------------------------------
// Kernel 3: expert FFN (SwiGLU) with bf16 WMMA, fused up/down projection
// grid: (N_TOK/16, N_EXP), block: 256 (8 waves)
// ---------------------------------------------------------------------------
__global__ __launch_bounds__(256)
void moe_ffn(const float* __restrict__ x,
             const float* __restrict__ w1, const float* __restrict__ w3,
             const float* __restrict__ w2,
             const int* __restrict__ cnt, const int* __restrict__ tok_list,
             const float* __restrict__ gate_list, float* __restrict__ out) {
  const int e = blockIdx.y;
  const int mt = blockIdx.x;
  const int count = cnt[e];
  if (mt * TILE_M >= count) return;

  __shared__ __bf16 xs[TILE_M][D_MODEL];   // 32 KB gathered activations
  __shared__ __bf16 hs[TILE_M][FCHUNK];    // 4 KB  SwiGLU staging
  __shared__ int    tok[TILE_M];
  __shared__ float  gv[TILE_M];

  const int tid  = threadIdx.x;
  const int wave = tid >> 5;
  const int lane = tid & 31;
  const int l    = lane & 15;
  const int hsel = lane >> 4;

  if (tid < TILE_M) {
    int idx = mt * TILE_M + tid;
    if (idx < count) {
      tok[tid] = tok_list[e * N_TOK + idx];
      gv[tid]  = gate_list[e * N_TOK + idx];
    } else {
      tok[tid] = 0;
      gv[tid]  = 0.0f;   // padded rows contribute nothing
    }
  }
  __syncthreads();

  // gather 16 token rows fp32 -> bf16 LDS (16 threads per row, 64 cols each)
  {
    int r  = tid >> 4;
    int c0 = (tid & 15) * 64;
    const float* src = x + (size_t)tok[r] * D_MODEL + c0;
#pragma unroll 4
    for (int i = 0; i < 64; i += 4) {
      float4 v = *(const float4*)(src + i);
      xs[r][c0 + i + 0] = (__bf16)v.x;
      xs[r][c0 + i + 1] = (__bf16)v.y;
      xs[r][c0 + i + 2] = (__bf16)v.z;
      xs[r][c0 + i + 3] = (__bf16)v.w;
    }
  }
  __syncthreads();

  // 64 persistent output accumulators: wave handles C-tiles c0=(j*8+wave)*16
  v8f acc[8] = {};

  const size_t ew = (size_t)e * D_FF * D_MODEL;
  const float* w1r = w1 + ew;                  // [D_FF][D_MODEL]
  const float* w3r = w3 + ew;
  const float* w2r = w2 + (size_t)e * D_MODEL * D_FF;  // [D_MODEL][D_FF]

  for (int fc = 0; fc < D_FF / FCHUNK; ++fc) {
    // ---- Phase A: h = silu(x*w1^T) * (x*w3^T) for this wave's 16-wide f tile
    const int f0 = fc * FCHUNK + wave * 16;
    v8f accg = {}, accu = {};
    const float* b1row = w1r + (size_t)(f0 + l) * D_MODEL + hsel * 16;
    const float* b3row = w3r + (size_t)(f0 + l) * D_MODEL + hsel * 16;
    for (int k0 = 0; k0 < D_MODEL; k0 += 32) {
      // prefetch ~1KB ahead in the streamed weight rows (global_prefetch_b8)
      __builtin_prefetch(b1row + k0 + 256, 0, 0);
      __builtin_prefetch(b3row + k0 + 256, 0, 0);
      v16bf a  = lds_a_frag(&xs[0][0], l, k0, D_MODEL, hsel);
      v16bf b1 = glb_b_frag_f32(b1row + k0);
      accg = __builtin_amdgcn_wmma_f32_16x16x32_bf16(false, a, false, b1,
                                                     (short)0, accg, false, false);
      v16bf b3 = glb_b_frag_f32(b3row + k0);
      accu = __builtin_amdgcn_wmma_f32_16x16x32_bf16(false, a, false, b3,
                                                     (short)0, accu, false, false);
    }

    __syncthreads();  // previous chunk's Phase-B reads of hs are done
#pragma unroll
    for (int r = 0; r < 8; ++r) {
      float g = accg[r];
      float u = accu[r];
      // silu(g)*u = g * sigmoid(g) * u ; fast v_rcp is safe (denom >= 1)
      float sig = __builtin_amdgcn_rcpf(1.0f + __expf(-g));
      float h = g * sig * u;
      int m = hsel ? (8 + r) : r;                // C/D row layout
      hs[m][wave * 16 + l] = (__bf16)h;
    }
    __syncthreads();  // hs visible to all waves

    // ---- Phase B: Y += h_chunk * w2^T  (K = FCHUNK slice of d_ff)
    for (int kk = 0; kk < FCHUNK; kk += 32) {
      v16bf a = lds_a_frag(&hs[0][0], l, kk, FCHUNK, hsel);
#pragma unroll
      for (int j = 0; j < 8; ++j) {
        int c0 = (j * 8 + wave) * 16;
        const float* b2row = w2r + (size_t)(c0 + l) * D_FF +
                             (size_t)fc * FCHUNK + kk + hsel * 16;
        __builtin_prefetch(b2row + 256, 0, 0);
        v16bf b = glb_b_frag_f32(b2row);
        acc[j] = __builtin_amdgcn_wmma_f32_16x16x32_bf16(false, a, false, b,
                                                         (short)0, acc[j], false, false);
      }
    }
  }

  // ---- scatter: out[token, c] += gate * y   (each token hit by its 2 experts)
#pragma unroll
  for (int j = 0; j < 8; ++j) {
    int c = (j * 8 + wave) * 16 + l;
#pragma unroll
    for (int r = 0; r < 8; ++r) {
      int m = hsel ? (8 + r) : r;
      float val = gv[m] * acc[j][r];
      unsafeAtomicAdd(&out[(size_t)tok[m] * D_MODEL + c], val);
    }
  }
}

// ---------------------------------------------------------------------------
// Host launch
// ---------------------------------------------------------------------------
extern "C" void kernel_launch(void* const* d_in, const int* in_sizes, int n_in,
                              void* d_out, int out_size, void* d_ws, size_t ws_size,
                              hipStream_t stream) {
  const float* x      = (const float*)d_in[0];   // [4,1024,1024]
  const float* gate_w = (const float*)d_in[1];   // [8,1024]
  const float* w1     = (const float*)d_in[2];   // [8,2816,1024]
  const float* w3     = (const float*)d_in[3];   // [8,2816,1024]
  const float* w2     = (const float*)d_in[4];   // [8,1024,2816]
  float* out = (float*)d_out;                    // [N_TOK*D_MODEL] ++ [loss]

  // workspace layout
  char* ws = (char*)d_ws;
  int*   cnt       = (int*)ws;                          // 8 ints
  float* rsum      = (float*)(ws + 32);                 // 8 floats
  int*   tok_list  = (int*)(ws + 256);                  // 8*4096 ints
  float* gate_list = (float*)(ws + 256 + N_EXP * N_TOK * sizeof(int));

  const int out_n = N_TOK * D_MODEL + 1;

  moe_init<<<1024, 256, 0, stream>>>(out, out_n, cnt, rsum);
  moe_gate<<<N_TOK / 256, 256, 0, stream>>>(x, gate_w, cnt, tok_list, gate_list, rsum);
  moe_loss<<<1, 1, 0, stream>>>(cnt, rsum, out + (size_t)N_TOK * D_MODEL);
  moe_ffn<<<dim3(N_TOK / TILE_M, N_EXP), 256, 0, stream>>>(
      x, w1, w3, w2, cnt, tok_list, gate_list, out);
}
